// PCFNet_8272107012335
// MI455X (gfx1250) — matrix operand order repaired and checked by reading, not compiled, not run on hardware
//
#include <hip/hip_runtime.h>

// PointNet++ MSG (ball query + per-point MLP + max pool) for MI455X / gfx1250.
// One 256-thread (8 wave32) workgroup per query point. MLP GEMMs run on
// v_wmma_f32_16x16x32_f16 with fused BN+ReLU epilogue and LDS atomic-max pool.

#define BQ 4
#define SQ 1024
#define NP 2048
#define NTHREADS 256
#define NWAVES 8
#define AST 136           // activation LDS row stride (halfs): 272B = 17*16B
#define EPSBN 1e-5f

typedef _Float16 half_t;
typedef _Float16 v16h __attribute__((ext_vector_type(16)));
typedef float    v8f  __attribute__((ext_vector_type(8)));

struct alignas(16) Q16 { unsigned u[4]; };

struct Args {
  const float* ori; const float* xyz;
  const float* w[9]; const float* bi[9]; const float* ga[9];
  const float* be[9]; const float* mn[9]; const float* vr[9];
  float* out;
};

__device__ __forceinline__ v16h ldfrag(const half_t* p0, const half_t* p1) {
  union { Q16 q[2]; v16h v; } u;
  u.q[0] = *reinterpret_cast<const Q16*>(p0);   // ds_load_b128
  u.q[1] = *reinterpret_cast<const Q16*>(p1);   // ds_load_b128
  return u.v;
}

__launch_bounds__(NTHREADS)
__global__ void pcf_msg_kernel(Args a) {
  constexpr int   KARR[3]  = {64, 64, 128};
  constexpr float RAD2[3]  = {0.04f*0.04f, 0.08f*0.08f, 0.16f*0.16f};
  constexpr int   CH[3][4] = {{3,32,32,64},{3,64,64,128},{3,64,96,128}};
  constexpr int   CHOFF[3] = {0, 64, 192};

  __shared__ alignas(16) half_t sA0[128*AST];   // activations ping
  __shared__ alignas(16) half_t sA1[128*AST];   // activations pong
  __shared__ alignas(16) half_t sW [128*104];   // weights, row = out-ch
  __shared__ float sSc[128], sBs[128];          // fused BN scale/shift
  __shared__ int   sIdx[3][128];                // ball-query indices
  __shared__ int   sMax[128];                   // max-pool (int image of f32>=0)

  const int tid  = threadIdx.x;
  const int wave = tid >> 5, lane = tid & 31;
  const int bsid = blockIdx.x;
  const int b = bsid / SQ, s = bsid % SQ;

  const float* xyzB = a.xyz + (size_t)b * 3 * NP;
  const float cx = a.ori[(b*SQ + s)*3 + 0];
  const float cy = a.ori[(b*SQ + s)*3 + 1];
  const float cz = a.ori[(b*SQ + s)*3 + 2];

  // output part 1: new_xyz transposed to [B,3,S]
  if (tid < 3) {
    float v = (tid == 0) ? cx : ((tid == 1) ? cy : cz);
    a.out[(b*3 + tid)*SQ + s] = v;
  }
  // warm the L2 path for this batch's cloud (global_prefetch_b8)
  if (wave == 3) __builtin_prefetch(xyzB + lane * (3*NP/32), 0, 0);

  // ---- ball query: wave w scans scale w in index order (wave32 compaction) ----
  if (wave < 3) {
    const float r2 = RAD2[wave];
    const int   K  = KARR[wave];
    int cnt = 0;
    for (int base = 0; base < NP && cnt < K; base += 32) {
      int n = base + lane;
      float dx = xyzB[n]        - cx;
      float dy = xyzB[NP + n]   - cy;
      float dz = xyzB[2*NP + n] - cz;
      bool in = (dx*dx + dy*dy + dz*dz) <= r2;
      unsigned m = __builtin_amdgcn_ballot_w32(in);
      int pos = cnt + __builtin_popcount(m & ((1u << lane) - 1u));
      if (in && pos < K) sIdx[wave][pos] = n;
      cnt += __builtin_popcount(m);
    }
    if (cnt < K) {
      // backfill with first hit; empty ball -> idx N, JAX clamps gather to N-1
      int fill = (cnt > 0) ? sIdx[wave][0] : (NP - 1);
      for (int k = cnt + lane; k < K; k += 32) sIdx[wave][k] = fill;
    }
  }
  __syncthreads();

#pragma unroll
  for (int sc3 = 0; sc3 < 3; ++sc3) {
    const int K = KARR[sc3];
    // gather center-relative coords -> sA0[K][0..2], zero-pad cols 3..31
    for (int k = tid; k < K; k += NTHREADS) {
      int n = sIdx[sc3][k];
      half_t* row = sA0 + k*AST;
      row[0] = (half_t)(xyzB[n]        - cx);
      row[1] = (half_t)(xyzB[NP + n]   - cy);
      row[2] = (half_t)(xyzB[2*NP + n] - cz);
#pragma unroll
      for (int c = 3; c < 32; ++c) row[c] = (half_t)0.f;
    }
    __syncthreads();

    half_t* aIn  = sA0;
    half_t* aOut = sA1;
#pragma unroll
    for (int ly = 0; ly < 3; ++ly) {
      const int  li   = sc3*3 + ly;
      const int  Cin  = CH[sc3][ly];
      const int  CinP = (Cin < 32) ? 32 : Cin;   // 32/64/96, all K-step multiples
      const int  Cout = CH[sc3][ly+1];
      const int  WST  = CinP + 8;                // bank-conflict pad, 16B-aligned rows
      const bool last = (ly == 2);
      const float* wp = a.w[li];

      // stage W (row-major per out-channel == B-fragment friendly), zero-pad K
      for (int i = tid; i < Cout*CinP; i += NTHREADS) {
        int o = i / CinP, c = i - o*CinP;
        sW[o*WST + c] = (c < Cin) ? (half_t)wp[o*Cin + c] : (half_t)0.f;
      }
      // fused inference-BN:  y = lin*sc + bs
      if (tid < Cout) {
        float sc = a.ga[li][tid] * rsqrtf(a.vr[li][tid] + EPSBN);
        sSc[tid] = sc;
        sBs[tid] = (a.bi[li][tid] - a.mn[li][tid]) * sc + a.be[li][tid];
      }
      if (last && tid < Cout) sMax[tid] = 0;     // ReLU output >= 0
      __syncthreads();

      const int Mt = K >> 4, Nt = Cout >> 4, Ks = CinP >> 5;
      const int l = lane & 15, hi = lane >> 4;
      for (int t = wave; t < Mt*Nt; t += NWAVES) {
        int mt = t / Nt, nt = t - mt*Nt;
        // A (16x32 f16): lane half 'hi' owns K in [hi*8,hi*8+8) and +16
        const half_t* arow = aIn + (mt*16 + l)*AST + hi*8;
        // B (32x16 f16): lane half 'hi' owns K in [hi*8,hi*8+8) and +16
        const half_t* brow = sW  + (nt*16 + l)*WST + hi*8;
        v8f acc = {0.f,0.f,0.f,0.f,0.f,0.f,0.f,0.f};
#pragma unroll
        for (int ks = 0; ks < Ks; ++ks) {
          v16h af = ldfrag(arow + ks*32, arow + ks*32 + 16);
          v16h bf = ldfrag(brow + ks*32, brow + ks*32 + 16);
          acc = __builtin_amdgcn_wmma_f32_16x16x32_f16(
              false, af, false, bf, (short)0, acc, false, false);
        }
        // D layout: VGPR r, lane(l,hi) -> row = mt*16 + hi*8 + r, ch = nt*16 + l
        int ch = nt*16 + l;
        float scv = sSc[ch], bsv = sBs[ch];
#pragma unroll
        for (int r = 0; r < 8; ++r) {
          float y = fmaxf(acc[r]*scv + bsv, 0.f);
          if (!last) {
            aOut[(mt*16 + hi*8 + r)*AST + ch] = (half_t)y;
          } else {
            atomicMax(&sMax[ch], __float_as_int(y)); // monotone for y >= 0
          }
        }
      }
      __syncthreads();
      half_t* tmp = aIn; aIn = aOut; aOut = tmp;
    }

    // output part 2: new_points [B,320,S]
    const int Cfin = CH[sc3][3];
    float* outP = a.out + (size_t)(BQ*3*SQ);
    for (int ch = tid; ch < Cfin; ch += NTHREADS) {
      outP[((size_t)b*320 + CHOFF[sc3] + ch)*SQ + s] = __int_as_float(sMax[ch]);
    }
    __syncthreads();
  }
}

extern "C" void kernel_launch(void* const* d_in, const int* in_sizes, int n_in,
                              void* d_out, int out_size, void* d_ws, size_t ws_size,
                              hipStream_t stream) {
  (void)d_ws; (void)ws_size; (void)out_size;
  Args a;
  a.out = (float*)d_out;
  int pOff;
  if (n_in >= 2 && in_sizes[1] == 3*NP*BQ / 1 && in_sizes[1] == 24576) {
    // insertion order: ori_xyz, xyz, params...
    a.ori = (const float*)d_in[0];
    a.xyz = (const float*)d_in[1];
    pOff = 2;
  } else {
    // sorted flatten: ori_xyz, params..., xyz
    a.ori = (const float*)d_in[0];
    a.xyz = (const float*)d_in[n_in - 1];
    pOff = 1;
  }
  // Within-layer key order: detect position of w (96 elems in layer 0).
  int wPos = 0;
  for (int j = 0; j < 6; ++j) {
    if (pOff + j < n_in && in_sizes[pOff + j] == 96) { wPos = j; break; }
  }
  int o_w, o_b, o_g, o_be, o_m, o_v;
  if (wPos == 0) { o_w = 0; o_b = 1; o_g = 2; o_be = 3; o_m = 4; o_v = 5; }   // insertion
  else           { o_b = 0; o_be = 1; o_g = 2; o_m = 3; o_v = 4; o_w = 5; }   // alphabetical
  for (int li = 0; li < 9; ++li) {
    const int base = pOff + li*6;
    a.w [li] = (const float*)d_in[base + o_w];
    a.bi[li] = (const float*)d_in[base + o_b];
    a.ga[li] = (const float*)d_in[base + o_g];
    a.be[li] = (const float*)d_in[base + o_be];
    a.mn[li] = (const float*)d_in[base + o_m];
    a.vr[li] = (const float*)d_in[base + o_v];
  }
  pcf_msg_kernel<<<dim3(BQ*SQ), dim3(NTHREADS), 0, stream>>>(a);
}
